// EGAttention_39719857554066
// MI455X (gfx1250) — matrix operand order repaired and checked by reading, not compiled, not run on hardware
//
#include <hip/hip_runtime.h>
#include <hip/hip_bf16.h>
#include <math.h>

typedef __attribute__((ext_vector_type(16))) __bf16 v16bf;
typedef __attribute__((ext_vector_type(8)))  float  v8f;

#define Bsz 4
#define Hh  128
#define Ww  128
#define Cc  256
#define NHh 8
#define HDd 32

#define QK_STR 36    // halfs per row, Q/K LDS [128][36]
#define VT_STR 136   // halfs per row, Vt LDS  [32][136]
#define AT_STR 136   // halfs per row, attn strip LDS [128][136]

__device__ __constant__ float kPI = 3.14159265358979323846f;

// Load Q,K (f32 global, row-major [128][32]) into bf16 LDS and build the
// RPE bias table. Table is indexed by i = q - k + 127 so that, per lane,
// consecutive accumulator rows j read consecutive addresses (vectorizable).
__device__ __forceinline__ void load_qk_bias(const float* __restrict__ qb,
                                             const float* __restrict__ kb,
                                             __bf16* Qs, __bf16* Ks,
                                             float* biasT, int h) {
    const int tid = threadIdx.x;
    for (int i = tid; i < Ww * HDd; i += 256) {
        const int pos = i >> 5;
        const int d   = i & 31;
        Qs[pos * QK_STR + d] = (__bf16)qb[(size_t)pos * Cc + d];
        Ks[pos * QK_STR + d] = (__bf16)kb[(size_t)pos * Cc + d];
    }
    const float rowscale = sinf((float)h * (kPI / 128.0f));
    for (int i = tid; i < 255; i += 256) {
        // i = q - k + 127  ->  base = (k - q) * 2pi/127 = (127 - i) * 2pi/127
        const float base = (float)(127 - i) * (2.0f * kPI / 127.0f);
        const float pn   = (base > 0.0f) ? 1.0f : -1.0f;
        const float mag  = sqrtf(fmaxf(2.0f * (1.0f - cosf(base)), 0.0f));
        biasT[i] = 0.1f * pn * rowscale * mag;   // 0.1 = BIAS_LEVEL folded in
    }
}

// One wave computes a 16x128 strip of attn = Q K^T + bias for its window/head.
// acc[t] is the 16x16 f32 tile at columns 16t..16t+15. The RPE bias enters as
// the WMMA C operand (no post-WMMA VALU adds, no RAW hazard stall).
__device__ __forceinline__ void compute_attn(const __bf16* Qs, const __bf16* Ks,
                                             const float* biasT,
                                             int r0, int lane, v8f* acc) {
    const int half = lane >> 4;
    const int m    = lane & 15;
    // A fragment: Q rows r0..r0+15, full K=32 (contiguous: element e -> k = 16*half + e)
    v16bf a;
    const __bf16* qrow = Qs + (r0 + m) * QK_STR;
#pragma unroll
    for (int e = 0; e < 8; ++e) a[e]     = qrow[16 * half + e];
#pragma unroll
    for (int e = 0; e < 8; ++e) a[8 + e] = qrow[16 * half + 8 + e];

#pragma unroll
    for (int t = 0; t < 8; ++t) {
        const int c0 = 16 * t;
        // B fragment: K^T tile (32x16); lane n+16h, element e -> Kmat[c0+n][16h+e]
        v16bf bb;
        const __bf16* krow = Ks + (c0 + m) * QK_STR + 16 * half;
#pragma unroll
        for (int e = 0; e < 16; ++e) bb[e] = krow[e];
        // C = bias: element j is (q = r0 + j + 8*half, k = c0 + m)
        // -> table index (q - k + 127) = (r0 + 8*half - c0 - m + 127) + j
        v8f c;
        const float* brow = biasT + (r0 + 8 * half - c0 - m + 127);
#pragma unroll
        for (int j = 0; j < 8; ++j) c[j] = brow[j];
        acc[t] = __builtin_amdgcn_wmma_f32_16x16x32_bf16(false, a, false, bb,
                                                         (short)0, c, false, false);
    }
}

__global__ __launch_bounds__(256) void eg_phaseA(const float* __restrict__ qkv,
                                                 float* __restrict__ Msum) {
    __shared__ __bf16 Qs[Ww * QK_STR];
    __shared__ __bf16 Ks[Ww * QK_STR];
    __shared__ float  biasT[255];
    __shared__ float  redbuf[8];

    const int w = blockIdx.x;          // window 0..511
    const int n = blockIdx.y;          // head 0..7
    const int b = w >> 7, h = w & 127;
    const size_t plane = (size_t)Bsz * Hh * Ww * Cc;
    const size_t base  = ((size_t)b * (Hh * Ww) + (size_t)h * Ww) * Cc + (size_t)n * HDd;

    load_qk_bias(qkv + base, qkv + plane + base, Qs, Ks, biasT, h);
    __syncthreads();

    const int lane = threadIdx.x & 31;
    const int wid  = threadIdx.x >> 5;
    v8f acc[8];
    compute_attn(Qs, Ks, biasT, 16 * wid, lane, acc);

    float s = 0.0f;
#pragma unroll
    for (int t = 0; t < 8; ++t)
#pragma unroll
        for (int j = 0; j < 8; ++j) s += fabsf(acc[t][j]);
#pragma unroll
    for (int off = 16; off >= 1; off >>= 1) s += __shfl_xor(s, off, 32);
    if (lane == 0) redbuf[wid] = s;
    __syncthreads();
    if (threadIdx.x == 0) {
        float tot = 0.0f;
        for (int i = 0; i < 8; ++i) tot += redbuf[i];
        atomicAdd(&Msum[w], tot);
    }
}

__global__ __launch_bounds__(512) void eg_max(float* __restrict__ Msum) {
    __shared__ float sm[512];
    sm[threadIdx.x] = Msum[threadIdx.x];
    __syncthreads();
    for (int s = 256; s > 0; s >>= 1) {
        if (threadIdx.x < s)
            sm[threadIdx.x] = fmaxf(sm[threadIdx.x], sm[threadIdx.x + s]);
        __syncthreads();
    }
    if (threadIdx.x == 0) Msum[512] = sm[0];
}

__global__ __launch_bounds__(256) void eg_phaseC(const float* __restrict__ qkv,
                                                 const float* __restrict__ res,
                                                 const float* __restrict__ Msum,
                                                 float* __restrict__ out) {
    __shared__ __bf16 Qs[Ww * QK_STR];
    __shared__ __bf16 Ks[Ww * QK_STR];
    __shared__ __bf16 Vt[HDd * VT_STR];       // transposed V: Vt[d][kpos]
    __shared__ __bf16 attnS[Ww * AT_STR];     // transformed attn, bf16
    __shared__ float  biasT[255];

    const int w = blockIdx.x;
    const int n = blockIdx.y;
    const int b = w >> 7, h = w & 127;
    const size_t plane = (size_t)Bsz * Hh * Ww * Cc;
    const size_t base  = ((size_t)b * (Hh * Ww) + (size_t)h * Ww) * Cc + (size_t)n * HDd;

    load_qk_bias(qkv + base, qkv + plane + base, Qs, Ks, biasT, h);
    {   // V loaded transposed so B-fragments are contiguous per lane
        const float* vb = qkv + 2 * plane + base;
        for (int i = threadIdx.x; i < Ww * HDd; i += 256) {
            const int pos = i >> 5;
            const int d   = i & 31;
            Vt[d * VT_STR + pos] = (__bf16)vb[(size_t)pos * Cc + d];
        }
    }
    __syncthreads();

    const int lane = threadIdx.x & 31;
    const int wid  = threadIdx.x >> 5;
    const int half = lane >> 4;
    const int m    = lane & 15;
    const int r0   = 16 * wid;

    v8f acc[8];
    compute_attn(Qs, Ks, biasT, r0, lane, acc);

    const float Mgate = fmaxf(Msum[w] / Msum[512], 0.5f);

    // Per-row L2 norm (row j lives across lanes of the same 16-lane half) then
    // cosine transform + gate.
#pragma unroll
    for (int j = 0; j < 8; ++j) {
        float s = 0.0f;
#pragma unroll
        for (int t = 0; t < 8; ++t) { const float x = acc[t][j]; s += x * x; }
        s += __shfl_xor(s, 1, 32);
        s += __shfl_xor(s, 2, 32);
        s += __shfl_xor(s, 4, 32);
        s += __shfl_xor(s, 8, 32);
        const float nrm   = fmaxf(sqrtf(s), 1e-12f);
        const float scale = (0.5f * kPI) / nrm;
#pragma unroll
        for (int t = 0; t < 8; ++t)
            acc[t][j] = (1.0f - cosf(acc[t][j] * scale)) * Mgate;
    }

    // Stage transformed attn strip as bf16 in LDS (C/D layout -> row major)
#pragma unroll
    for (int t = 0; t < 8; ++t)
#pragma unroll
        for (int j = 0; j < 8; ++j)
            attnS[(r0 + j + 8 * half) * AT_STR + 16 * t + m] = (__bf16)acc[t][j];
    __syncthreads();

    // x = attn @ V : 16x32 result per wave, K = 128 in 4 chained WMMA steps
    v8f z = {};
    v8f o[2] = {z, z};
    const __bf16* arow = attnS + (r0 + m) * AT_STR;
#pragma unroll
    for (int kk = 0; kk < 4; ++kk) {
        const int kb0 = kk * 32;
        v16bf a;
#pragma unroll
        for (int e = 0; e < 8; ++e) a[e]     = arow[kb0 + 16 * half + e];
#pragma unroll
        for (int e = 0; e < 8; ++e) a[8 + e] = arow[kb0 + 16 * half + 8 + e];
#pragma unroll
        for (int t = 0; t < 2; ++t) {
            v16bf bb;
            const __bf16* vrow = Vt + (16 * t + m) * VT_STR + kb0 + 16 * half;
#pragma unroll
            for (int e = 0; e < 16; ++e) bb[e] = vrow[e];
            o[t] = __builtin_amdgcn_wmma_f32_16x16x32_bf16(false, a, false, bb,
                                                           (short)0, o[t], false, false);
        }
    }

    // out = x + res * (1 - M)
    const float rfac = 1.0f - Mgate;
#pragma unroll
    for (int t = 0; t < 2; ++t)
#pragma unroll
        for (int j = 0; j < 8; ++j) {
            const int q = r0 + j + 8 * half;
            const int d = 16 * t + m;
            const size_t idx = base + (size_t)q * Cc + d;
            out[idx] = o[t][j] + res[idx] * rfac;
        }
}

extern "C" void kernel_launch(void* const* d_in, const int* in_sizes, int n_in,
                              void* d_out, int out_size, void* d_ws, size_t ws_size,
                              hipStream_t stream) {
    const float* qkv = (const float*)d_in[0];
    const float* res = (const float*)d_in[1];
    float* out  = (float*)d_out;
    float* Msum = (float*)d_ws;               // [512] window sums + [1] global max

    hipMemsetAsync(Msum, 0, 513 * sizeof(float), stream);
    dim3 grid(Bsz * Hh, NHh, 1);              // (512, 8)
    eg_phaseA<<<grid, 256, 0, stream>>>(qkv, Msum);
    eg_max<<<1, 512, 0, stream>>>(Msum);
    eg_phaseC<<<grid, 256, 0, stream>>>(qkv, res, Msum, out);
}
